// MultiQueryAttention_37520834298283
// MI455X (gfx1250) — matrix operand order repaired and checked by reading, not compiled, not run on hardware
//
#include <hip/hip_runtime.h>

typedef __attribute__((ext_vector_type(16))) _Float16 v16h;
typedef __attribute__((ext_vector_type(8)))  _Float16 v8h;
typedef __attribute__((ext_vector_type(8)))  float    v8f;

#define WMMA_F32_F16(a, b, c) \
  __builtin_amdgcn_wmma_f32_16x16x32_f16(false, (a), false, (b), (short)0, (c), false, false)

// Async global -> LDS copy of 16 bytes (CDNA5 GLOBAL_LOAD_ASYNC_TO_LDS_B128).
// VDST = per-lane LDS byte address, VADDR = 64-bit global address, no SADDR.
// Tracked by ASYNCcnt; consumer must s_wait_asynccnt before reading LDS.
__device__ inline void async_copy_b128(const _Float16* g, _Float16* l) {
  asm volatile("global_load_async_to_lds_b128 %0, %1, off"
               :: "v"((unsigned int)(unsigned long long)l),
                  "v"((unsigned long long)g)
               : "memory");
}
__device__ inline void wait_async0() {
  asm volatile("s_wait_asynccnt 0" ::: "memory");
}

// ---------------------------------------------------------------------------
// Elementwise conversion kernels
// ---------------------------------------------------------------------------
__global__ void cvt_f32_to_f16(const float* __restrict__ in, _Float16* __restrict__ out, int n) {
  int i = blockIdx.x * blockDim.x + threadIdx.x;
  int stride = gridDim.x * blockDim.x;
  for (; i < n; i += stride) out[i] = (_Float16)in[i];
}

// Wt[n*K + k] = (f16) W[k*N + n]   (row-major [K,N] -> transposed f16 [N,K])
__global__ void transpose_cvt(const float* __restrict__ W, _Float16* __restrict__ Wt, int K, int N) {
  int i = blockIdx.x * blockDim.x + threadIdx.x;
  int total = K * N;
  int stride = gridDim.x * blockDim.x;
  for (; i < total; i += stride) {
    int k = i / N;
    int n = i - k * N;
    Wt[(size_t)n * K + k] = (_Float16)W[i];
  }
}

// ---------------------------------------------------------------------------
// WMMA fragment helpers (this code's 16-bit layouts: contiguous k, same for A and B)
// A 16x32: lane holds row m=lane&15; g=lane>>4: contiguous K = [16g .. 16g+15]
// B 32x16 (from K-major transposed storage Bt[n][k]): lane holds col n=lane&15,
//         contiguous K = [16g .. 16g+15]
// C/D 16x16 f32: lane group g holds rows r+8g, col = lane&15, VGPR r = row offset
// ---------------------------------------------------------------------------
__device__ inline v16h load_a_frag(const _Float16* base, int row_stride, int lane) {
  int m = lane & 15, g = lane >> 4;
  const _Float16* p = base + (size_t)m * row_stride + 16 * g;
  v16h a;
  *((v8h*)&a)       = *(const v8h*)(p);
  *(((v8h*)&a) + 1) = *(const v8h*)(p + 8);
  return a;
}
__device__ inline v16h load_b_frag(const _Float16* baseT, int row_stride, int lane) {
  int n = lane & 15, g = lane >> 4;
  return *(const v16h*)(baseT + (size_t)n * row_stride + 16 * g);
}

// ---------------------------------------------------------------------------
// Tiled WMMA GEMM: out = A[M,K](f16) x W[K,N] + bias, W given transposed (Wt[N,K], f16)
// BM=128, BK=64, BN in {128,64}. 256 threads = 8 waves.
//   BN=128: waves 2x4, wave tile 64x32 -> 4x2 WMMA tiles, 16 WMMA / K-step
//   BN= 64: waves 4x2, wave tile 32x32 -> 2x2 WMMA tiles,  8 WMMA / K-step
// Double-buffered LDS staged with async global->LDS copies (ASYNCcnt):
// stage(k+1) is issued, compute(k) runs from the other buffer, then one
// s_wait_asynccnt 0 + barrier per K-step. global_prefetch of step k+2.
// MODE 0: f16 row-major out [M,N]
// MODE 1: f32 row-major out [M,N]
// MODE 2: f16 batched-transposed out: out[b][n][s], m = b*batchS + s  (for V)
// ---------------------------------------------------------------------------
template <int MODE, int BN>
__global__ __launch_bounds__(256) void gemm_wmma(const _Float16* __restrict__ A,
                                                 const _Float16* __restrict__ Wt,
                                                 const float* __restrict__ bias,
                                                 void* __restrict__ outp,
                                                 int M, int N, int K, int batchS) {
  constexpr int BM = 128, BK = 64;
  constexpr int WN = BN / 32;          // wave columns (4 or 2)
  constexpr int WR = 8 / WN;           // wave rows (2 or 4)
  constexpr int TM = BM / WR;          // wave tile rows (64 or 32)
  constexpr int MT = TM / 16;          // m-subtiles per wave (4 or 2)
  constexpr int ACH = BM * BK / 8;     // 8-element (16 B) chunks in A tile
  constexpr int BCH = BN * BK / 8;
  constexpr int CPR = BK / 8;          // chunks per row

  __shared__ _Float16 sA[2][BM * BK];
  __shared__ _Float16 sB[2][BN * BK];

  const int tid = threadIdx.x;
  const int lane = tid & 31;
  const int widx = tid >> 5;
  const int wn = widx & (WN - 1);
  const int wm = widx / WN;
  const int m0 = blockIdx.x * BM;
  const int n0 = blockIdx.y * BN;

  v8f acc[MT][2] = {};

  auto stage = [&](int buf, int k0) {
    for (int i = 0; i < ACH / 256; i++) {
      int cid = tid + i * 256;
      int r = cid / CPR, c = (cid % CPR) * 8;
      async_copy_b128(A + (size_t)(m0 + r) * K + k0 + c, sA[buf] + r * BK + c);
    }
    for (int i = 0; i < BCH / 256; i++) {
      int cid = tid + i * 256;
      int r = cid / CPR, c = (cid % CPR) * 8;
      async_copy_b128(Wt + (size_t)(n0 + r) * K + k0 + c, sB[buf] + r * BK + c);
    }
  };

  stage(0, 0);
  wait_async0();
  __syncthreads();

  int buf = 0;
  for (int k0 = 0; k0 < K; k0 += BK) {
    if (k0 + BK < K) {
      stage(buf ^ 1, k0 + BK);
      if (k0 + 2 * BK < K) {
        __builtin_prefetch(A + (size_t)(m0 + (tid >> 1)) * K + k0 + 2 * BK, 0, 1);
        if (tid < 2 * BN)
          __builtin_prefetch(Wt + (size_t)(n0 + (tid >> 1)) * K + k0 + 2 * BK, 0, 1);
      }
    }
    for (int ks = 0; ks < BK; ks += 32) {
      v16h af[MT], bf[2];
      for (int mt = 0; mt < MT; mt++)
        af[mt] = load_a_frag(sA[buf] + (wm * TM + mt * 16) * BK + ks, BK, lane);
      for (int nt = 0; nt < 2; nt++)
        bf[nt] = load_b_frag(sB[buf] + (wn * 32 + nt * 16) * BK + ks, BK, lane);
      for (int mt = 0; mt < MT; mt++)
        for (int nt = 0; nt < 2; nt++)
          acc[mt][nt] = WMMA_F32_F16(af[mt], bf[nt], acc[mt][nt]);
    }
    wait_async0();
    __syncthreads();
    buf ^= 1;
  }

  // Epilogue: bias add + store per C/D layout
  const int nloc = lane & 15;
  const int hi = lane >> 4;
  for (int mt = 0; mt < MT; mt++) {
    for (int nt = 0; nt < 2; nt++) {
      int n_g = n0 + wn * 32 + nt * 16 + nloc;
      float bv = bias[n_g];
      for (int r = 0; r < 8; r++) {
        int m_g = m0 + wm * TM + mt * 16 + r + 8 * hi;
        float v = acc[mt][nt][r] + bv;
        if constexpr (MODE == 0) {
          ((_Float16*)outp)[(size_t)m_g * N + n_g] = (_Float16)v;
        } else if constexpr (MODE == 1) {
          ((float*)outp)[(size_t)m_g * N + n_g] = v;
        } else {
          int b = m_g / batchS, s = m_g - b * batchS;
          ((_Float16*)outp)[((size_t)b * N + n_g) * batchS + s] = (_Float16)v;
        }
      }
    }
  }
}

// ---------------------------------------------------------------------------
// Flash MQA attention: one wave handles 16 query rows of one head over all keys.
// q: f16 [B*S, E] (head h at column h*64), k: f16 [B*S, 64], vT: f16 [B, 64, S]
// ctx: f16 [B*S, E]. 1/sqrt(D)=0.125 is folded into the Q fragment (exact in f16).
// ---------------------------------------------------------------------------
__global__ __launch_bounds__(256) void mqa_attn(const _Float16* __restrict__ q,
                                                const _Float16* __restrict__ kk,
                                                const _Float16* __restrict__ vT,
                                                _Float16* __restrict__ ctx) {
  constexpr int S = 2048, D = 64, E = 1024;
  __shared__ _Float16 pbuf[8 * 16 * 32];  // 1 KB per wave
  const int lane = threadIdx.x & 31;
  const int widx = threadIdx.x >> 5;
  const int w = blockIdx.x * 8 + widx;    // 0 .. B*H*(S/16)-1 = 8191
  const int qt = w & 127;                 // query tile within (b,h)
  const int h = (w >> 7) & 15;
  const int b = w >> 11;
  const int s0 = qt * 16;
  const int m = lane & 15, g = lane >> 4;

  const _Float16* qbase = q + ((size_t)(b * S + s0)) * E + h * D;
  const _Float16* kbase = kk + (size_t)b * S * D;
  const _Float16* vbase = vT + (size_t)b * D * S;
  _Float16* pw = pbuf + widx * (16 * 32);

  // Q A-fragments for d-blocks [0,32) and [32,64), pre-scaled by 1/sqrt(D)
  v16h aQ[2];
  for (int db = 0; db < 2; db++) {
    const _Float16* p = qbase + (size_t)m * E + db * 32 + 16 * g;
    *((v8h*)&aQ[db])       = *(const v8h*)(p);
    *(((v8h*)&aQ[db]) + 1) = *(const v8h*)(p + 8);
    aQ[db] = aQ[db] * (_Float16)0.125f;
  }

  float mrow[8], lrow[8];
  v8f acc[4] = {};
  for (int r = 0; r < 8; r++) { mrow[r] = -3.0e38f; lrow[r] = 0.f; }

  for (int j0 = 0; j0 < S; j0 += 32) {
    // Scores: two 16x16 tiles, K-dim = 64 in two WMMAs each
    v8f sc[2] = {};
    for (int t = 0; t < 2; t++) {
      int j = j0 + t * 16 + m;  // this lane's key column
      const _Float16* kp = kbase + (size_t)j * D;
      v16h b0 = *(const v16h*)(kp + 16 * g);
      v16h b1 = *(const v16h*)(kp + 32 + 16 * g);
      sc[t] = WMMA_F32_F16(aQ[0], b0, sc[t]);
      sc[t] = WMMA_F32_F16(aQ[1], b1, sc[t]);
    }
    // Online softmax stats: rows live across 16-lane groups
    float corr[8];
    for (int r = 0; r < 8; r++) {
      float v = fmaxf(sc[0][r], sc[1][r]);
      v = fmaxf(v, __shfl_xor(v, 1));
      v = fmaxf(v, __shfl_xor(v, 2));
      v = fmaxf(v, __shfl_xor(v, 4));
      v = fmaxf(v, __shfl_xor(v, 8));
      float mn = fmaxf(mrow[r], v);
      corr[r] = __expf(mrow[r] - mn);
      mrow[r] = mn;
    }
    // P = exp(s - m): write to LDS in A-layout source form, accumulate row sums
    for (int r = 0; r < 8; r++) {
      float p0 = __expf(sc[0][r] - mrow[r]);
      float p1 = __expf(sc[1][r] - mrow[r]);
      int row = r + 8 * g;
      pw[row * 32 + m]      = (_Float16)p0;
      pw[row * 32 + 16 + m] = (_Float16)p1;
      float ps = p0 + p1;
      ps += __shfl_xor(ps, 1);
      ps += __shfl_xor(ps, 2);
      ps += __shfl_xor(ps, 4);
      ps += __shfl_xor(ps, 8);
      lrow[r] = lrow[r] * corr[r] + ps;
    }
    for (int nt = 0; nt < 4; nt++)
      for (int r = 0; r < 8; r++) acc[nt][r] *= corr[r];

    asm volatile("s_wait_dscnt 0" ::: "memory");  // wave-local LDS visibility

    // P A-fragment (16x32 over this key chunk)
    v16h aP;
    {
      const _Float16* p = pw + m * 32 + 16 * g;
      *((v8h*)&aP)       = *(const v8h*)(p);
      *(((v8h*)&aP) + 1) = *(const v8h*)(p + 8);
    }
    // PV: 4 output d-tiles, B-frags contiguous from transposed V
    for (int nt = 0; nt < 4; nt++) {
      int d = nt * 16 + m;
      v16h bv = *(const v16h*)(vbase + (size_t)d * S + j0 + 16 * g);
      acc[nt] = WMMA_F32_F16(aP, bv, acc[nt]);
    }
  }

  // Normalize and store context
  for (int nt = 0; nt < 4; nt++) {
    int d = nt * 16 + m;
    for (int r = 0; r < 8; r++) {
      int row = r + 8 * g;
      float v = acc[nt][r] / lrow[r];
      ctx[((size_t)(b * S + s0 + row)) * E + h * D + d] = (_Float16)v;
    }
  }
}

// ---------------------------------------------------------------------------
// Host launcher
// ---------------------------------------------------------------------------
extern "C" void kernel_launch(void* const* d_in, const int* in_sizes, int n_in,
                              void* d_out, int out_size, void* d_ws, size_t ws_size,
                              hipStream_t stream) {
  (void)in_sizes; (void)n_in; (void)out_size; (void)ws_size;
  const float* x  = (const float*)d_in[0];
  const float* Wq = (const float*)d_in[1];
  const float* bq = (const float*)d_in[2];
  const float* Wk = (const float*)d_in[3];
  const float* bk = (const float*)d_in[4];
  const float* Wv = (const float*)d_in[5];
  const float* bv = (const float*)d_in[6];
  const float* Wo = (const float*)d_in[7];
  const float* bo = (const float*)d_in[8];
  float* out = (float*)d_out;

  const int B = 4, S = 2048, E = 1024, D = 64, M = B * S;

  char* ws = (char*)d_ws;
  size_t off = 0;
  auto take = [&](size_t elems) -> _Float16* {
    _Float16* p = (_Float16*)(ws + off);
    off += ((elems * 2 + 255) & ~(size_t)255);
    return p;
  };
  _Float16* x16  = take((size_t)M * E);      // x in f16
  _Float16* Wqt  = take((size_t)E * E);      // Wq^T
  _Float16* Wkt  = take((size_t)D * E);      // Wk^T
  _Float16* Wvt  = take((size_t)D * E);      // Wv^T
  _Float16* Wot  = take((size_t)E * E);      // Wo^T
  _Float16* q16  = take((size_t)M * E);      // Q
  _Float16* k16  = take((size_t)M * D);      // K
  _Float16* vT16 = take((size_t)B * D * S);  // V transposed per batch
  _Float16* c16  = take((size_t)M * E);      // attention context

  cvt_f32_to_f16<<<2048, 256, 0, stream>>>(x, x16, M * E);
  transpose_cvt<<<1024, 256, 0, stream>>>(Wq, Wqt, E, E);
  transpose_cvt<<<256, 256, 0, stream>>>(Wk, Wkt, E, D);
  transpose_cvt<<<256, 256, 0, stream>>>(Wv, Wvt, E, D);
  transpose_cvt<<<1024, 256, 0, stream>>>(Wo, Wot, E, E);

  gemm_wmma<0, 128><<<dim3(M / 128, E / 128), 256, 0, stream>>>(x16, Wqt, bq, q16, M, E, E, S);
  gemm_wmma<0, 64><<<dim3(M / 128, 1), 256, 0, stream>>>(x16, Wkt, bk, k16, M, D, E, S);
  gemm_wmma<2, 64><<<dim3(M / 128, 1), 256, 0, stream>>>(x16, Wvt, bv, vT16, M, D, E, S);

  mqa_attn<<<(M / 16) / 8, 256, 0, stream>>>(q16, k16, vT16, c16);

  gemm_wmma<1, 128><<<dim3(M / 128, E / 128), 256, 0, stream>>>(c16, Wot, bo, out, M, E, E, S);
}
